// SparseConvDetector_32865089749521
// MI455X (gfx1250) — matrix operand order repaired and checked by reading, not compiled, not run on hardware
//
#include <hip/hip_runtime.h>
#include <hip/hip_bf16.h>
#include <math.h>

typedef __attribute__((ext_vector_type(16))) _Float16 v16h;
typedef __attribute__((ext_vector_type(8)))  float    v8f;

#define BN_INV_F 0.99999500003749981f  /* 1/sqrt(1+1e-5) */
#define GD 48
#define G3 (GD*GD*GD)
#define BATCH 2
#define NPTS 120000
#define VOXF 0.1f

// ---------------------------------------------------------------------------
// 1) Voxelize: scatter point sums + counts with float atomics
// ---------------------------------------------------------------------------
__global__ void voxelize_kernel(const float* __restrict__ pts,
                                float* __restrict__ sums, float* __restrict__ cnt) {
    int idx = blockIdx.x * blockDim.x + threadIdx.x;
    if (idx >= BATCH * NPTS) return;
    int b = idx / NPTS;
    const float* p = pts + (size_t)idx * 4;
    float x = p[0], y = p[1], z = p[2], w = p[3];
    int cx = min(max((int)floorf(x / VOXF), 0), GD - 1);
    int cy = min(max((int)floorf(y / VOXF), 0), GD - 1);
    int cz = min(max((int)floorf(z / VOXF), 0), GD - 1);
    int vid = (cx * GD + cy) * GD + cz;
    float* s = sums + ((size_t)b * G3 + vid) * 4;
    atomicAdd(s + 0, x);
    atomicAdd(s + 1, y);
    atomicAdd(s + 2, z);
    atomicAdd(s + 3, w);
    atomicAdd(cnt + (size_t)b * G3 + vid, 1.0f);
}

// ---------------------------------------------------------------------------
// 2) VFE: per-voxel mean -> MLP(4->32->16, BN+ReLU) -> f16 activations + mask
//    Output padded to 32 channels (upper 16 zero) so conv0 needs no K check.
// ---------------------------------------------------------------------------
__global__ void vfe_kernel(const float* __restrict__ sums, const float* __restrict__ cnt,
                           const float* __restrict__ w1, const float* __restrict__ b1,
                           const float* __restrict__ g1, const float* __restrict__ bt1,
                           const float* __restrict__ w2, const float* __restrict__ b2,
                           const float* __restrict__ g2, const float* __restrict__ bt2,
                           _Float16* __restrict__ x0, float* __restrict__ m0) {
    int idx = blockIdx.x * blockDim.x + threadIdx.x;
    if (idx >= BATCH * G3) return;
    float c = cnt[idx];
    float d = fmaxf(c, 1.0f);
    const float* s = sums + (size_t)idx * 4;
    float mean[4];
    #pragma unroll
    for (int i = 0; i < 4; ++i) mean[i] = s[i] / d;

    float h1[32];
    #pragma unroll
    for (int o = 0; o < 32; ++o) {
        float a = b1[o];
        #pragma unroll
        for (int i = 0; i < 4; ++i) a += mean[i] * w1[i * 32 + o];
        a = a * (g1[o] * BN_INV_F) + bt1[o];
        h1[o] = fmaxf(a, 0.0f);
    }
    float mk = (c > 0.0f) ? 1.0f : 0.0f;
    m0[idx] = mk;
    #pragma unroll
    for (int o = 0; o < 16; ++o) {
        float a = b2[o];
        #pragma unroll
        for (int k = 0; k < 32; ++k) a += h1[k] * w2[k * 16 + o];
        a = a * (g2[o] * BN_INV_F) + bt2[o];
        x0[(size_t)idx * 32 + o] = (_Float16)(fmaxf(a, 0.0f) * mk);
    }
    #pragma unroll
    for (int o = 16; o < 32; ++o) x0[(size_t)idx * 32 + o] = (_Float16)0.0f;
}

// ---------------------------------------------------------------------------
// 3) Weight repack: f32 [27][Ci][Co] -> f16 packed so each lane's WMMA
//    B-fragment (8 dwords) is contiguous:  u32 layout [t][kcb][g][co][i]
//    where ci = kcb*32 + g*16 + i*2 + h  (K pad to Cip).
// ---------------------------------------------------------------------------
__global__ void convert_weights(const float* __restrict__ w, _Float16* __restrict__ out,
                                int Ci, int Cip, int Co) {
    int idx = blockIdx.x * blockDim.x + threadIdx.x;
    int total = 27 * Cip * Co;
    if (idx >= total) return;
    int co = idx % Co;
    int r  = idx / Co;
    int ci = r % Cip;
    int t  = r / Cip;
    float v = (ci < Ci) ? w[((size_t)t * Ci + ci) * Co + co] : 0.0f;
    int nkc = Cip >> 5;
    int kcb = ci >> 5;
    int rr  = ci & 31;
    int g   = rr >> 4;
    int i   = (rr & 15) >> 1;
    int h   = ci & 1;
    size_t u32idx = ((((size_t)(t * nkc + kcb) * 2 + g) * Co + co) * 8 + i);
    out[u32idx * 2 + h] = (_Float16)v;
}

// ---------------------------------------------------------------------------
// 4) Implicit-GEMM 3D conv (pad 1) on v_wmma_f32_16x16x32_f16.
//    One wave per 16-position x 16-channel tile.  All 27 halo-selected tap
//    base offsets are precomputed into registers (27 cndmasks total, no
//    per-iteration exec masking); A and B fragments each = 2x
//    global_load_b128; 3-buffer / 2-stage software pipeline keeps two load
//    groups in flight past each WMMA.
// ---------------------------------------------------------------------------
template <int DI, int DO, int STRIDE, int CIP, int CO>
__global__ __launch_bounds__(32)
void conv3d_wmma(const _Float16* __restrict__ xin, const _Float16* __restrict__ wbuf,
                 _Float16* __restrict__ xout,
                 const float* __restrict__ g, const float* __restrict__ bt,
                 const float* __restrict__ mask, int zrel4) {
    constexpr int DSPI = DI * DI * DI;
    constexpr int DSP  = DO * DO * DO;
    constexpr int NKC  = CIP / 32;          // K chunks of 32
    constexpr int A4   = CIP / 8;           // uint4 per spatial position
    constexpr int NIT  = 27 * NKC;          // flattened (tap, kchunk) space

    const int lane = threadIdx.x;
    const int hi   = lane >> 4;
    const int lm   = lane & 15;
    const int bi   = blockIdx.z;
    const int tile = blockIdx.x;
    const int ch   = blockIdx.y * 16 + lm;   // output channel (B/D column)

    int m_a = tile * 16 + lm;                // A row: M = lm
    bool arow_ok = (m_a < DSP);
    int mm = arow_ok ? m_a : 0;
    int oz = mm / (DO * DO);
    int rr = mm - oz * DO * DO;
    int oy = rr / DO;
    int ox = rr - oy * DO;

    const uint4* xin4 = (const uint4*)xin;
    const uint4* w4   = (const uint4*)wbuf;
    const int    abase = bi * DSPI * A4;     // fits 32-bit (buffers < 16 MB)

    // Precompute halo-selected base offsets (uint4 units) for all 27 taps.
    int aoff[27];
    #pragma unroll
    for (int t = 0; t < 27; ++t) {
        int dz = t / 9, dy = (t / 3) % 3, dx = t % 3;
        int iz = oz * STRIDE + dz - 1;
        int iy = oy * STRIDE + dy - 1;
        int ix = ox * STRIDE + dx - 1;
        bool ok = arow_ok && ((unsigned)iz < (unsigned)DI)
                          && ((unsigned)iy < (unsigned)DI)
                          && ((unsigned)ix < (unsigned)DI);
        int ipos = (iz * DI + iy) * DI + ix;
        int aidx = abase + ipos * A4;
        aoff[t] = ok ? aidx : zrel4;         // v_cndmask, hoisted out of loop
    }

    struct Frag { uint4 a0, a1, b0, b1; };
    auto ld = [&](int it, Frag& f) {
        int t = it / NKC, kcb = it % NKC;
        int aidx = aoff[t] + kcb * 4;        // zero page sized for kcb reach
        // A 16x32 f16: lanes0-15 K=0..15, lanes16-31 K=16..31
        f.a0 = xin4[aidx + 2 * hi];
        f.a1 = xin4[aidx + 2 * hi + 1];
        // B 32x16 f16: lanes0-15 K=0..15, lanes16-31 K=16..31 (packed contig)
        int bidx = ((it * 2 + hi) * CO + ch) * 2;
        f.b0 = w4[bidx];
        f.b1 = w4[bidx + 1];
        if (kcb == 0 && t < 26)   // warm next tap's weights into cache
            __builtin_prefetch(wbuf + (size_t)(t + 1) * NKC * 2 * CO * 16, 0, 1);
    };

    v8f acc = {0.f, 0.f, 0.f, 0.f, 0.f, 0.f, 0.f, 0.f};
    Frag f[3];
    ld(0, f[0]);
    ld(1, f[1]);
    #pragma unroll
    for (int it = 0; it < NIT; ++it) {
        if (it + 2 < NIT) ld(it + 2, f[(it + 2) % 3]);  // 2 stages ahead
        Frag& cur = f[it % 3];
        union { v16h v; uint4 q[2]; } A, Bf;
        A.q[0]  = cur.a0;  A.q[1]  = cur.a1;
        Bf.q[0] = cur.b0;  Bf.q[1] = cur.b1;
        acc = __builtin_amdgcn_wmma_f32_16x16x32_f16(
            false, A.v, false, Bf.v, (short)0, acc, false, false);
    }

    // Epilogue: BN + ReLU + mask, f32 -> f16.  C/D: vgpr r -> M=r+8*hi, N=lm.
    float sc = g[ch] * BN_INV_F;
    float sh = bt[ch];
    #pragma unroll
    for (int r = 0; r < 8; ++r) {
        int mo = tile * 16 + (hi << 3) + r;
        if (mo < DSP) {
            float v = fmaxf(acc[r] * sc + sh, 0.0f) * mask[bi * DSP + mo];
            xout[(size_t)(bi * DSP + mo) * CO + ch] = (_Float16)v;
        }
    }
}

// ---------------------------------------------------------------------------
// 5) Mask max-pool 3x3x3 stride 2 pad 1
// ---------------------------------------------------------------------------
__global__ void pool_mask_kernel(const float* __restrict__ min_, float* __restrict__ mout,
                                 int Di, int Do, int total) {
    int idx = blockIdx.x * blockDim.x + threadIdx.x;
    if (idx >= total) return;
    int Dsp = Do * Do * Do;
    int b = idx / Dsp;
    int m = idx - b * Dsp;
    int oz = m / (Do * Do);
    int r  = m - oz * Do * Do;
    int oy = r / Do;
    int ox = r - oy * Do;
    float mx = 0.0f;
    #pragma unroll
    for (int dz = 0; dz < 3; ++dz)
        for (int dy = 0; dy < 3; ++dy)
            for (int dx = 0; dx < 3; ++dx) {
                int iz = 2 * oz + dz - 1, iy = 2 * oy + dy - 1, ix = 2 * ox + dx - 1;
                if ((unsigned)iz < (unsigned)Di && (unsigned)iy < (unsigned)Di &&
                    (unsigned)ix < (unsigned)Di)
                    mx = fmaxf(mx, min_[((long)b * Di * Di * Di) + ((long)iz * Di + iy) * Di + ix]);
            }
    mout[idx] = mx;
}

// ---------------------------------------------------------------------------
// 6) Heads: mean pool, cls/box MLPs, sigmoid score, corner decode
// ---------------------------------------------------------------------------
__global__ void head_kernel(const _Float16* __restrict__ x4,
                            const float* cw1, const float* cb1, const float* cg, const float* cbt,
                            const float* cw2, const float* cb2,
                            const float* bw1, const float* bb1, const float* bg, const float* bbt,
                            const float* bw2, const float* bb2,
                            float* __restrict__ out) {
    __shared__ float feats[256];
    __shared__ float hc[128];
    __shared__ float hb[128];
    __shared__ float boxv[7];
    int b = blockIdx.x;
    int t = threadIdx.x;

    float s = 0.0f;
    for (int k = 0; k < 27; ++k) s += (float)x4[((long)b * 27 + k) * 256 + t];
    feats[t] = s * (1.0f / 27.0f);
    __syncthreads();

    if (t < 128) {
        float a = cb1[t], d = bb1[t];
        for (int k = 0; k < 256; ++k) {
            float f = feats[k];
            a += f * cw1[k * 128 + t];
            d += f * bw1[k * 128 + t];
        }
        a = a * (cg[t] * BN_INV_F) + cbt[t];
        d = d * (bg[t] * BN_INV_F) + bbt[t];
        hc[t] = fmaxf(a, 0.0f);
        hb[t] = fmaxf(d, 0.0f);
    }
    __syncthreads();

    if (t < 7) {
        float a = bb2[t];
        for (int k = 0; k < 128; ++k) a += hb[k] * bw2[k * 7 + t];
        boxv[t] = a;
    }
    if (t == 0) {
        float a = cb2[0];
        for (int k = 0; k < 128; ++k) a += hc[k] * cw2[k];
        out[b] = 1.0f / (1.0f + expf(-a));
    }
    __syncthreads();

    if (t == 0) {
        float px = boxv[0], py = boxv[1], pz = boxv[2];
        float ex = expf(boxv[3]), ey = expf(boxv[4]), ez = expf(boxv[5]);
        float chh = cosf(boxv[6]), shh = sinf(boxv[6]);
        #pragma unroll
        for (int k = 0; k < 8; ++k) {
            float ux = ((k >> 2) & 1) ? 0.5f : -0.5f;
            float uy = ((k >> 1) & 1) ? 0.5f : -0.5f;
            float uz = (k & 1) ? 0.5f : -0.5f;
            float sx = ux * ex, sy = uy * ey, sz = uz * ez;
            float cx = sx * chh - sy * shh;
            float cy = sx * shh + sy * chh;
            float cz = sz;
            float* o = out + 2 + (long)b * 24 + k * 3;
            o[0] = cx + px; o[1] = cy + py; o[2] = cz + pz;
        }
    }
}

// ---------------------------------------------------------------------------
// Launch
// ---------------------------------------------------------------------------
extern "C" void kernel_launch(void* const* d_in, const int* in_sizes, int n_in,
                              void* d_out, int out_size, void* d_ws, size_t ws_size,
                              hipStream_t stream) {
    const float* points  = (const float*)d_in[0];
    const float* vfe_w1  = (const float*)d_in[1];
    const float* vfe_b1  = (const float*)d_in[2];
    const float* vfe_g1  = (const float*)d_in[3];
    const float* vfe_bt1 = (const float*)d_in[4];
    const float* vfe_w2  = (const float*)d_in[5];
    const float* vfe_b2  = (const float*)d_in[6];
    const float* vfe_g2  = (const float*)d_in[7];
    const float* vfe_bt2 = (const float*)d_in[8];
    const float* bk1[4], *bg1[4], *bbt1[4], *bk2[4], *bg2[4], *bbt2[4];
    for (int i = 0; i < 4; ++i) {
        bk1[i]  = (const float*)d_in[9 + 6 * i + 0];
        bg1[i]  = (const float*)d_in[9 + 6 * i + 1];
        bbt1[i] = (const float*)d_in[9 + 6 * i + 2];
        bk2[i]  = (const float*)d_in[9 + 6 * i + 3];
        bg2[i]  = (const float*)d_in[9 + 6 * i + 4];
        bbt2[i] = (const float*)d_in[9 + 6 * i + 5];
    }
    const float* cls_w1 = (const float*)d_in[33];
    const float* cls_b1 = (const float*)d_in[34];
    const float* cls_g  = (const float*)d_in[35];
    const float* cls_bt = (const float*)d_in[36];
    const float* cls_w2 = (const float*)d_in[37];
    const float* cls_b2 = (const float*)d_in[38];
    const float* box_w1 = (const float*)d_in[39];
    const float* box_b1 = (const float*)d_in[40];
    const float* box_g  = (const float*)d_in[41];
    const float* box_bt = (const float*)d_in[42];
    const float* box_w2 = (const float*)d_in[43];
    const float* box_b2 = (const float*)d_in[44];

    // ----- workspace carve-up -----
    char* ws = (char*)d_ws;
    size_t off = 0;
    auto alloc = [&](size_t bytes) -> size_t {
        size_t o = off;
        off = (off + bytes + 255) & ~(size_t)255;
        return o;
    };
    const int D0 = 48, D1 = 24, D2 = 12, D3 = 6, D4 = 3;
    const int S0 = D0*D0*D0, S1 = D1*D1*D1, S2 = D2*D2*D2, S3 = D3*D3*D3, S4 = D4*D4*D4;

    size_t o_sums = alloc((size_t)BATCH * G3 * 4 * sizeof(float));
    size_t o_cnt  = alloc((size_t)BATCH * G3 * sizeof(float));
    size_t o_zero = alloc(1024);  // zero page for halo reads (kcb reach incl.)
    size_t atom_bytes = (o_zero + 1024) - o_sums;

    size_t o_m0 = alloc((size_t)BATCH * S0 * sizeof(float));
    size_t o_m1 = alloc((size_t)BATCH * S1 * sizeof(float));
    size_t o_m2 = alloc((size_t)BATCH * S2 * sizeof(float));
    size_t o_m3 = alloc((size_t)BATCH * S3 * sizeof(float));
    size_t o_m4 = alloc((size_t)BATCH * S4 * sizeof(float));

    size_t o_x0 = alloc((size_t)BATCH * S0 * 32  * 2);   // padded to 32 ch
    size_t o_t0 = alloc((size_t)BATCH * S0 * 32  * 2);
    size_t o_x1 = alloc((size_t)BATCH * S1 * 32  * 2);
    size_t o_t1 = alloc((size_t)BATCH * S1 * 64  * 2);
    size_t o_x2 = alloc((size_t)BATCH * S2 * 64  * 2);
    size_t o_t2 = alloc((size_t)BATCH * S2 * 128 * 2);
    size_t o_x3 = alloc((size_t)BATCH * S3 * 128 * 2);
    size_t o_t3 = alloc((size_t)BATCH * S3 * 256 * 2);
    size_t o_x4 = alloc((size_t)BATCH * S4 * 256 * 2);

    size_t o_w[8];
    const int wCi[8]  = {16, 32, 32, 64, 64, 128, 128, 256};
    const int wCip[8] = {32, 32, 32, 64, 64, 128, 128, 256};
    const int wCo[8]  = {32, 32, 64, 64, 128, 128, 256, 256};
    for (int i = 0; i < 8; ++i)
        o_w[i] = alloc((size_t)27 * wCip[i] * wCo[i] * 2);

    float*    sums = (float*)(ws + o_sums);
    float*    cnt  = (float*)(ws + o_cnt);
    float*    m[5] = {(float*)(ws + o_m0), (float*)(ws + o_m1), (float*)(ws + o_m2),
                      (float*)(ws + o_m3), (float*)(ws + o_m4)};
    _Float16* x0 = (_Float16*)(ws + o_x0);
    _Float16* t0 = (_Float16*)(ws + o_t0);
    _Float16* x1 = (_Float16*)(ws + o_x1);
    _Float16* t1 = (_Float16*)(ws + o_t1);
    _Float16* x2 = (_Float16*)(ws + o_x2);
    _Float16* t2 = (_Float16*)(ws + o_t2);
    _Float16* x3 = (_Float16*)(ws + o_x3);
    _Float16* t3 = (_Float16*)(ws + o_t3);
    _Float16* x4 = (_Float16*)(ws + o_x4);
    _Float16* wbuf[8];
    for (int i = 0; i < 8; ++i) wbuf[i] = (_Float16*)(ws + o_w[i]);
    // zero-page offset relative to each conv input, in uint4 (16 B) units
    auto zrel = [&](size_t o_x) -> int { return (int)(((long)o_zero - (long)o_x) / 16); };

    // ----- pipeline -----
    hipMemsetAsync(ws + o_sums, 0, atom_bytes, stream);  // sums + cnt + zero page

    voxelize_kernel<<<(BATCH * NPTS + 255) / 256, 256, 0, stream>>>(points, sums, cnt);

    vfe_kernel<<<(BATCH * G3 + 255) / 256, 256, 0, stream>>>(
        sums, cnt, vfe_w1, vfe_b1, vfe_g1, vfe_bt1,
        vfe_w2, vfe_b2, vfe_g2, vfe_bt2, x0, m[0]);

    const float* wsrc[8] = {bk1[0], bk2[0], bk1[1], bk2[1], bk1[2], bk2[2], bk1[3], bk2[3]};
    for (int i = 0; i < 8; ++i) {
        int total = 27 * wCip[i] * wCo[i];
        convert_weights<<<(total + 255) / 256, 256, 0, stream>>>(
            wsrc[i], wbuf[i], wCi[i], wCip[i], wCo[i]);
    }

    auto launch_pool = [&](const float* mi, float* mo, int Di, int Do) {
        int total = BATCH * Do * Do * Do;
        pool_mask_kernel<<<(total + 255) / 256, 256, 0, stream>>>(mi, mo, Di, Do, total);
    };

    // Block 0:  48^3 x(16pad32) -> 48^3 x32 -> 24^3 x32
    conv3d_wmma<48,48,1,32,32><<<dim3((S0+15)/16, 2, BATCH), 32, 0, stream>>>(
        x0, wbuf[0], t0, bg1[0], bbt1[0], m[0], zrel(o_x0));
    launch_pool(m[0], m[1], D0, D1);
    conv3d_wmma<48,24,2,32,32><<<dim3((S1+15)/16, 2, BATCH), 32, 0, stream>>>(
        t0, wbuf[1], x1, bg2[0], bbt2[0], m[1], zrel(o_t0));
    // Block 1:  24^3 x32 -> 24^3 x64 -> 12^3 x64
    conv3d_wmma<24,24,1,32,64><<<dim3((S1+15)/16, 4, BATCH), 32, 0, stream>>>(
        x1, wbuf[2], t1, bg1[1], bbt1[1], m[1], zrel(o_x1));
    launch_pool(m[1], m[2], D1, D2);
    conv3d_wmma<24,12,2,64,64><<<dim3((S2+15)/16, 4, BATCH), 32, 0, stream>>>(
        t1, wbuf[3], x2, bg2[1], bbt2[1], m[2], zrel(o_t1));
    // Block 2:  12^3 x64 -> 12^3 x128 -> 6^3 x128
    conv3d_wmma<12,12,1,64,128><<<dim3((S2+15)/16, 8, BATCH), 32, 0, stream>>>(
        x2, wbuf[4], t2, bg1[2], bbt1[2], m[2], zrel(o_x2));
    launch_pool(m[2], m[3], D2, D3);
    conv3d_wmma<12,6,2,128,128><<<dim3((S3+15)/16, 8, BATCH), 32, 0, stream>>>(
        t2, wbuf[5], x3, bg2[2], bbt2[2], m[3], zrel(o_t2));
    // Block 3:  6^3 x128 -> 6^3 x256 -> 3^3 x256
    conv3d_wmma<6,6,1,128,256><<<dim3((S3+15)/16, 16, BATCH), 32, 0, stream>>>(
        x3, wbuf[6], t3, bg1[3], bbt1[3], m[3], zrel(o_x3));
    launch_pool(m[3], m[4], D3, D4);
    conv3d_wmma<6,3,2,256,256><<<dim3((S4+15)/16, 16, BATCH), 32, 0, stream>>>(
        t3, wbuf[7], x4, bg2[3], bbt2[3], m[4], zrel(o_t3));

    head_kernel<<<BATCH, 256, 0, stream>>>(
        x4, cls_w1, cls_b1, cls_g, cls_bt, cls_w2, cls_b2,
        box_w1, box_b1, box_g, box_bt, box_w2, box_b2, (float*)d_out);
}
